// LeNet5_48378511622209
// MI455X (gfx1250) — 8-bit matrix operand order repaired and checked by reading, compiled, not run on hardware
//
#include <hip/hip_runtime.h>


typedef int v8i __attribute__((ext_vector_type(8)));

static constexpr int BATCH = 16384;

static constexpr size_t OFF_W1M  = 0;
static constexpr size_t OFF_W2B  = 256;
static constexpr size_t OFF_WF1  = OFF_W2B + 3072;
static constexpr size_t OFF_WF2  = OFF_WF1 + 57344;
static constexpr size_t OFF_WF3  = OFF_WF2 + 16384;
static constexpr size_t OFF_A1   = OFF_WF3 + 2048;
static constexpr size_t OFF_C2   = OFF_A1 + (size_t)BATCH * 1176;
static constexpr size_t OFF_A2   = OFF_C2 + (size_t)BATCH * 1600;
static constexpr size_t OFF_A3   = OFF_A2 + (size_t)BATCH * 448;
static constexpr size_t OFF_A4   = OFF_A3 + (size_t)BATCH * 128;

__device__ __forceinline__ signed char sgnf(float v) {
    return (signed char)((v > 0.0f) - (v < 0.0f));
}
__device__ __forceinline__ signed char sgni(int v) {
    return (signed char)((v > 0) - (v < 0));
}
__device__ __forceinline__ int load_i32(const signed char* p) {
    return *(const int*)p;
}
__device__ __forceinline__ int kA(int v, int g) { return 32 * (v >> 2) + 16 * g + 4 * (v & 3); }

__global__ __launch_bounds__(256) void prep_kernel(
    const float* __restrict__ w1, const float* __restrict__ w2,
    const float* __restrict__ wfc1, const float* __restrict__ wfc2,
    const float* __restrict__ wfc3,
    unsigned int* __restrict__ w1m, signed char* __restrict__ w2b,
    signed char* __restrict__ wf1b, signed char* __restrict__ wf2b,
    signed char* __restrict__ wf3b)
{
    int t = blockIdx.x * 256 + threadIdx.x;
    constexpr int S1 = 6;
    constexpr int S2 = 16 * 192;
    constexpr int S3 = 128 * 448;
    constexpr int S4 = 128 * 128;
    constexpr int S5 = 16 * 128;
    if (t < S1) {
        unsigned int m = 0;
        for (int j = 0; j < 25; ++j)
            m |= (unsigned int)(w1[t * 25 + j] > 0.0f) << j;
        w1m[t] = m;
    } else if (t < S1 + S2) {
        int i = t - S1, oc = i / 192, k = i % 192;
        w2b[i] = (k < 150) ? sgnf(w2[oc * 150 + k]) : (signed char)0;
    } else if (t < S1 + S2 + S3) {
        int i = t - S1 - S2, n = i / 448, k = i % 448;
        wf1b[i] = (n < 120 && k < 400) ? sgnf(wfc1[n * 400 + k]) : (signed char)0;
    } else if (t < S1 + S2 + S3 + S4) {
        int i = t - S1 - S2 - S3, n = i / 128, k = i % 128;
        wf2b[i] = (n < 84 && k < 120) ? sgnf(wfc2[n * 120 + k]) : (signed char)0;
    } else if (t < S1 + S2 + S3 + S4 + S5) {
        int i = t - S1 - S2 - S3 - S4, n = i / 128, k = i % 128;
        wf3b[i] = (n < 10 && k < 84) ? sgnf(wfc3[n * 84 + k]) : (signed char)0;
    }
}

__global__ __launch_bounds__(256) void conv1_kernel(
    const float* __restrict__ x, const unsigned int* __restrict__ w1m,
    signed char* __restrict__ a1)
{
    int t = blockIdx.x * 256 + threadIdx.x;
    int px = t % 14, py = (t / 14) % 14, oc = (t / 196) % 6, b = t / 1176;

    const float* xb = x + (size_t)b * 1024 + (2 * py) * 32 + 2 * px;
    __builtin_prefetch(xb + 64, 0, 0);

    unsigned int row[6];
    #pragma unroll
    for (int dy = 0; dy < 6; ++dy) {
        unsigned int rb = 0;
        #pragma unroll
        for (int dx = 0; dx < 6; ++dx)
            rb |= (unsigned int)(xb[dy * 32 + dx] > 0.0f) << dx;
        row[dy] = rb;
    }

    unsigned int wm = w1m[oc];

    int best = -1000;
    #pragma unroll
    for (int wy = 0; wy < 2; ++wy)
        #pragma unroll
        for (int wx = 0; wx < 2; ++wx) {
            unsigned int patch = 0;
            #pragma unroll
            for (int ky = 0; ky < 5; ++ky)
                patch |= ((row[wy + ky] >> wx) & 31u) << (ky * 5);
            int sum = 25 - 2 * __builtin_popcount(patch ^ wm);
            best = max(best, sum);
        }
    a1[t] = sgni(best);
}

__device__ __forceinline__ unsigned int gath_dword(
    const signed char* img, const int roff[30], int d)
{
    unsigned int w = 0;
    #pragma unroll
    for (int j = 0; j < 4; ++j) {
        int k = d * 4 + j;
        if (k < 150) {
            int rowi = (k / 25) * 5 + (k % 25) / 5;
            int cc   = k % 5;
            w |= (unsigned int)(unsigned char)img[roff[rowi] + cc] << (8 * j);
        }
    }
    return w;
}

__global__ __launch_bounds__(256) void conv2_kernel(
    const signed char* __restrict__ a1, const signed char* __restrict__ w2b,
    signed char* __restrict__ c2)
{
    int wave = (blockIdx.x * 256 + threadIdx.x) >> 5;
    wave = __builtin_amdgcn_readfirstlane(wave);
    int lane = threadIdx.x & 31;
    int g = lane >> 4, nl = lane & 15;

    int bimg = wave / 7, sub = wave % 7;
    int p = sub * 16 + nl;
    bool valid = p < 100;
    int pc = valid ? p : 0;
    int oy = pc / 10, ox = pc % 10;

    const signed char* img = a1 + (size_t)bimg * 1176;

    int roff[30];
    #pragma unroll
    for (int ic = 0; ic < 6; ++ic)
        #pragma unroll
        for (int rr = 0; rr < 5; ++rr)
            roff[ic * 5 + rr] = (ic * 14 + oy + rr) * 14 + ox;

    const signed char* wrow = w2b + nl * 192;

    v8i acc = {0, 0, 0, 0, 0, 0, 0, 0};
    #pragma unroll
    for (int s = 0; s < 3; ++s) {
        v8i a, bm;
        #pragma unroll
        for (int v = 0; v < 8; ++v)
            a[v] = load_i32(wrow + s * 64 + kA(v, g));
        #pragma unroll
        for (int v = 0; v < 8; ++v) {
            int i0 = 16 * s + 8 * (v >> 2) + (v & 3);
            unsigned int lo = gath_dword(img, roff, i0);
            unsigned int hi = gath_dword(img, roff, i0 + 4);
            bm[v] = (int)(g ? hi : lo);
        }
        acc = __builtin_amdgcn_wmma_i32_16x16x64_iu8(true, a, true, bm, acc, false, false);
    }
    #pragma unroll
    for (int r = 0; r < 8; ++r) {
        if (valid) {
            int oc = r + 8 * g;
            c2[((size_t)bimg * 16 + oc) * 100 + pc] = (signed char)acc[r];
        }
    }
}

__global__ __launch_bounds__(256) void pool2_kernel(
    const signed char* __restrict__ c2, signed char* __restrict__ a2)
{
    int t = blockIdx.x * 256 + threadIdx.x;
    int j = t % 448, b = t / 448;
    signed char r = 0;
    if (j < 400) {
        int oc = j / 25, q = j % 25, py = q / 5, px = q % 5;
        const signed char* c = c2 + ((size_t)b * 16 + oc) * 100;
        int i0 = (2 * py) * 10 + 2 * px;
        int m = max(max((int)c[i0], (int)c[i0 + 1]), max((int)c[i0 + 10], (int)c[i0 + 11]));
        r = sgni(m);
    }
    a2[t] = r;
}

template <int KPAD, int NTILES, int NOUT>
__global__ __launch_bounds__(256) void fc_sign_kernel(
    const signed char* __restrict__ act, const signed char* __restrict__ wgt,
    signed char* __restrict__ out)
{
    int wave = (blockIdx.x * 256 + threadIdx.x) >> 5;
    wave = __builtin_amdgcn_readfirstlane(wave);
    int lane = threadIdx.x & 31;
    int g = lane >> 4, nl = lane & 15;
    int mtile = wave / NTILES, ntile = wave % NTILES;

    const signed char* arow = act + (size_t)(mtile * 16 + nl) * KPAD;
    const signed char* wrow = wgt + (size_t)(ntile * 16 + nl) * KPAD;

    v8i acc = {0, 0, 0, 0, 0, 0, 0, 0};
    #pragma unroll
    for (int s = 0; s < KPAD / 64; ++s) {
        v8i a, bm;
        #pragma unroll
        for (int v = 0; v < 8; ++v) {
            a[v]  = load_i32(arow + s * 64 + kA(v, g));
            bm[v] = load_i32(wrow + s * 64 + 32 * (v >> 2) + 16 * g + 4 * (v & 3));
        }
        acc = __builtin_amdgcn_wmma_i32_16x16x64_iu8(true, a, true, bm, acc, false, false);
    }
    #pragma unroll
    for (int r = 0; r < 8; ++r) {
        int m = mtile * 16 + r + 8 * g;
        int n = ntile * 16 + nl;
        out[(size_t)m * (NTILES * 16) + n] = (n < NOUT) ? sgni(acc[r]) : (signed char)0;
    }
}

__global__ __launch_bounds__(256) void fc3_kernel(
    const signed char* __restrict__ act, const signed char* __restrict__ wgt,
    float* __restrict__ out)
{
    int wave = (blockIdx.x * 256 + threadIdx.x) >> 5;
    wave = __builtin_amdgcn_readfirstlane(wave);
    int lane = threadIdx.x & 31;
    int g = lane >> 4, nl = lane & 15;

    const signed char* arow = act + (size_t)(wave * 16 + nl) * 128;
    const signed char* wrow = wgt + (size_t)nl * 128;

    v8i acc = {0, 0, 0, 0, 0, 0, 0, 0};
    #pragma unroll
    for (int s = 0; s < 2; ++s) {
        v8i a, bm;
        #pragma unroll
        for (int v = 0; v < 8; ++v) {
            a[v]  = load_i32(arow + s * 64 + kA(v, g));
            bm[v] = load_i32(wrow + s * 64 + 32 * (v >> 2) + 16 * g + 4 * (v & 3));
        }
        acc = __builtin_amdgcn_wmma_i32_16x16x64_iu8(true, a, true, bm, acc, false, false);
    }
    #pragma unroll
    for (int r = 0; r < 8; ++r) {
        int m = wave * 16 + r + 8 * g;
        int n = nl;
        if (n < 10) out[(size_t)m * 10 + n] = (float)acc[r];
    }
}

extern "C" void kernel_launch(void* const* d_in, const int* in_sizes, int n_in,
                              void* d_out, int out_size, void* d_ws, size_t ws_size,
                              hipStream_t stream) {
    const float* x    = (const float*)d_in[0];
    const float* w1   = (const float*)d_in[1];
    const float* w2   = (const float*)d_in[2];
    const float* wfc1 = (const float*)d_in[3];
    const float* wfc2 = (const float*)d_in[4];
    const float* wfc3 = (const float*)d_in[5];
    float* out = (float*)d_out;

    char* ws = (char*)d_ws;
    unsigned int* w1m = (unsigned int*)(ws + OFF_W1M);
    signed char* w2b  = (signed char*)(ws + OFF_W2B);
    signed char* wf1b = (signed char*)(ws + OFF_WF1);
    signed char* wf2b = (signed char*)(ws + OFF_WF2);
    signed char* wf3b = (signed char*)(ws + OFF_WF3);
    signed char* a1   = (signed char*)(ws + OFF_A1);
    signed char* c2   = (signed char*)(ws + OFF_C2);
    signed char* a2   = (signed char*)(ws + OFF_A2);
    signed char* a3   = (signed char*)(ws + OFF_A3);
    signed char* a4   = (signed char*)(ws + OFF_A4);

    prep_kernel<<<309, 256, 0, stream>>>(w1, w2, wfc1, wfc2, wfc3,
                                         w1m, w2b, wf1b, wf2b, wf3b);
    conv1_kernel<<<75264, 256, 0, stream>>>(x, w1m, a1);
    conv2_kernel<<<14336, 256, 0, stream>>>(a1, w2b, c2);
    pool2_kernel<<<28672, 256, 0, stream>>>(c2, a2);
    fc_sign_kernel<448, 8, 120><<<1024, 256, 0, stream>>>(a2, wf1b, a3);
    fc_sign_kernel<128, 8, 84><<<1024, 256, 0, stream>>>(a3, wf2b, a4);
    fc3_kernel<<<128, 256, 0, stream>>>(a4, wf3b, out);
}
